// GraphSAGEEncoder_6743098655467
// MI455X (gfx1250) — matrix operand order repaired and checked by reading, not compiled, not run on hardware
//
#include <hip/hip_runtime.h>
#include <hip/hip_bf16.h>

#define GN 50000
#define GE 800000
#define GD 128
#define LN_EPS 1e-5f
#define APAD 136   // LDS row stride (halfwords) for bf16 A tiles, keeps 16B align

typedef __attribute__((ext_vector_type(16))) __bf16       v16bf;
typedef __attribute__((ext_vector_type(8)))  float        v8f;
typedef __attribute__((ext_vector_type(8)))  unsigned int v8u;

// ---------- helpers ----------
__device__ __forceinline__ unsigned short f2bf_u(float f) {
    unsigned u = __builtin_bit_cast(unsigned, f);
    return (unsigned short)((u + 0x7FFFu + ((u >> 16) & 1u)) >> 16); // RNE
}

union BCast { v8u v; uint4 q[2]; };

// A-fragment from LDS bf16 tile (row-major, APAD stride).
// lane: row = lane%16, half = lane/16 -> K runs {base..base+15} (base = kt*32 + half*16)
__device__ __forceinline__ v16bf load_a_lds(const unsigned short* s,
                                            int row, int kt, int h16) {
    const unsigned short* p = s + row * APAD + kt * 32 + h16 * 16;
    BCast b;
    b.q[0] = *(const uint4*)(p);        // 8 halves  (ds_load_b128)
    b.q[1] = *(const uint4*)(p + 8);    // 8 halves  (ds_load_b128)
    return __builtin_bit_cast(v16bf, b.v);
}

// B-fragment: weights pre-stored transposed [n][k] bf16 in global (L2-resident);
// lane (column n, half h16) reads 16 contiguous bf16 = 32 bytes.
__device__ __forceinline__ v16bf load_b_frag(const unsigned short* __restrict__ wT,
                                             int n, int kbase) {
    v8u r = *(const v8u*)(wT + (size_t)n * GD + kbase);
    return __builtin_bit_cast(v16bf, r);
}

// ---------- kernels ----------
__global__ void zero_f(float* __restrict__ p, long n) {
    long i = (long)blockIdx.x * blockDim.x + threadIdx.x;
    if (i < n) p[i] = 0.0f;
}

// convert fp32 W[k][n] -> bf16 WT[n][k]
__global__ void prep_w(const float* __restrict__ W, unsigned short* __restrict__ wT) {
    int t = blockIdx.x * 256 + threadIdx.x;   // 0 .. 16383
    int k = t >> 7, n = t & 127;
    wT[(size_t)n * GD + k] = f2bf_u(W[t]);
}

__global__ void count_deg(const int* __restrict__ dst, float* __restrict__ cnt) {
    int e = blockIdx.x * blockDim.x + threadIdx.x;
    if (e < GE)
        __hip_atomic_fetch_add(&cnt[dst[e]], 1.0f,
                               __ATOMIC_RELAXED, __HIP_MEMORY_SCOPE_AGENT);
}

// one edge handled by 32 threads, 4 floats each -> native fp32 global atomics
__global__ void scatter_add(const float* __restrict__ h,
                            const int* __restrict__ src,
                            const int* __restrict__ dst,
                            float* __restrict__ agg) {
    long t = (long)blockIdx.x * blockDim.x + threadIdx.x;
    long e = t >> 5;
    if (e >= GE) return;
    int c = (int)(t & 31) * 4;
    int s = src[e], d = dst[e];
    float4 v = *(const float4*)(h + (size_t)s * GD + c);
    float* p = agg + (size_t)d * GD + c;
    __hip_atomic_fetch_add(p + 0, v.x, __ATOMIC_RELAXED, __HIP_MEMORY_SCOPE_AGENT);
    __hip_atomic_fetch_add(p + 1, v.y, __ATOMIC_RELAXED, __HIP_MEMORY_SCOPE_AGENT);
    __hip_atomic_fetch_add(p + 2, v.z, __ATOMIC_RELAXED, __HIP_MEMORY_SCOPE_AGENT);
    __hip_atomic_fetch_add(p + 3, v.w, __ATOMIC_RELAXED, __HIP_MEMORY_SCOPE_AGENT);
}

// Fused: out16x128 = (agg/cnt) @ Wl + h @ Wr + b, then LayerNorm + ReLU.
// Block = 256 threads = 8 waves; wave w owns output columns [16w, 16w+16).
__global__ __launch_bounds__(256)
void sage_gemm_ln(const float* __restrict__ h, const float* __restrict__ agg,
                  const float* __restrict__ cnt,
                  const unsigned short* __restrict__ wlT,
                  const unsigned short* __restrict__ wrT,
                  const float* __restrict__ bias, const float* __restrict__ gam,
                  const float* __restrict__ bet, float* __restrict__ out) {
    __shared__ unsigned short sH[16 * APAD];   // bf16 h tile
    __shared__ unsigned short sM[16 * APAD];   // bf16 (agg/cnt) tile
    __shared__ float          sInv[16];
    __shared__ float          sOut[16 * 132];

    const int tid  = threadIdx.x;
    const int wave = tid >> 5;
    const int lane = tid & 31;
    const int h16  = lane >> 4;
    const int l16  = lane & 15;
    const int m0   = blockIdx.x * 16;

    // per-row reciprocal degree (single v_rcp_f32, no IEEE div chain)
    if (tid < 16)
        sInv[tid] = __builtin_amdgcn_rcpf(fmaxf(cnt[m0 + tid], 1.0f));
    __syncthreads();

    // cooperative staging: 16x128 h and mean tiles -> LDS bf16 (8 elems each /mat)
#pragma unroll
    for (int i = 0; i < 8; ++i) {
        int idx = tid + 256 * i;               // 0..2047
        int row = idx >> 7, col = idx & 127;
        size_t g = (size_t)(m0 + row) * GD + col;
        sH[row * APAD + col] = f2bf_u(h[g]);
        sM[row * APAD + col] = f2bf_u(agg[g] * sInv[row]);
    }
    __syncthreads();

    const int   ncol = wave * 16 + l16;
    const float bn   = bias[ncol];

    v8f c = {};
#pragma unroll
    for (int kt = 0; kt < 4; ++kt) {
        const int kbase = kt * 32 + h16 * 16;
        v16bf ah = load_a_lds(sH, l16, kt, h16);
        v16bf am = load_a_lds(sM, l16, kt, h16);
        v16bf br = load_b_frag(wrT, ncol, kbase);
        v16bf bl = load_b_frag(wlT, ncol, kbase);
        c = __builtin_amdgcn_wmma_f32_16x16x32_bf16(false, ah, false, br,
                                                    (short)0, c, false, false);
        c = __builtin_amdgcn_wmma_f32_16x16x32_bf16(false, am, false, bl,
                                                    (short)0, c, false, false);
    }

    // C layout: VGPR i, lane -> row = i + 8*h16, col = ncol
#pragma unroll
    for (int i = 0; i < 8; ++i)
        sOut[(i + 8 * h16) * 132 + ncol] = c[i] + bn;

    __syncthreads();

    // LayerNorm + ReLU: 16 threads per row, 8 elements each
    const int rr = tid >> 4;
    const int j  = tid & 15;
    float v[8], s = 0.0f, ss = 0.0f;
#pragma unroll
    for (int i = 0; i < 8; ++i) {
        float t = sOut[rr * 132 + j + 16 * i];
        v[i] = t; s += t; ss += t * t;
    }
#pragma unroll
    for (int m = 8; m >= 1; m >>= 1) {
        s  += __shfl_xor(s,  m, 16);
        ss += __shfl_xor(ss, m, 16);
    }
    const float mu  = s * (1.0f / 128.0f);
    const float var = ss * (1.0f / 128.0f) - mu * mu;
    const float rs  = rsqrtf(var + LN_EPS);

    float* orow = out + (size_t)(m0 + rr) * GD;
#pragma unroll
    for (int i = 0; i < 8; ++i) {
        int   ci = j + 16 * i;
        float y  = (v[i] - mu) * rs * gam[ci] + bet[ci];
        orow[ci] = fmaxf(y, 0.0f);
    }
}

// ---------- launcher ----------
extern "C" void kernel_launch(void* const* d_in, const int* in_sizes, int n_in,
                              void* d_out, int out_size, void* d_ws, size_t ws_size,
                              hipStream_t stream) {
    const float* x   = (const float*)d_in[0];
    const int*   ei  = (const int*)d_in[1];
    const float* Wl1 = (const float*)d_in[2];
    const float* Wr1 = (const float*)d_in[3];
    const float* b1  = (const float*)d_in[4];
    const float* g1  = (const float*)d_in[5];
    const float* be1 = (const float*)d_in[6];
    const float* Wl2 = (const float*)d_in[7];
    const float* Wr2 = (const float*)d_in[8];
    const float* b2  = (const float*)d_in[9];
    const float* g2  = (const float*)d_in[10];
    const float* be2 = (const float*)d_in[11];
    float*       out = (float*)d_out;

    const int* src = ei;        // edge_index[0]
    const int* dst = ei + GE;   // edge_index[1]

    // workspace carve-up
    float* agg = (float*)d_ws;
    float* cnt = agg + (size_t)GN * GD;
    float* h1  = cnt + GN;
    unsigned short* wlT1 = (unsigned short*)(h1 + (size_t)GN * GD);
    unsigned short* wrT1 = wlT1 + GD * GD;
    unsigned short* wlT2 = wrT1 + GD * GD;
    unsigned short* wrT2 = wlT2 + GD * GD;

    const long aggN = (long)GN * GD;
    const int  zB   = (int)((aggN + 255) / 256);
    const int  zC   = (GN + 255) / 256;
    const int  eB   = (GE + 255) / 256;
    const long sT   = (long)GE * 32;
    const int  sB   = (int)((sT + 255) / 256);
    const int  gB   = GN / 16;           // 3125

    // weights -> bf16 transposed (L2-resident thereafter)
    prep_w<<<64, 256, 0, stream>>>(Wl1, wlT1);
    prep_w<<<64, 256, 0, stream>>>(Wr1, wrT1);
    prep_w<<<64, 256, 0, stream>>>(Wl2, wlT2);
    prep_w<<<64, 256, 0, stream>>>(Wr2, wrT2);

    // degrees (shared by both layers)
    zero_f<<<zC, 256, 0, stream>>>(cnt, GN);
    count_deg<<<eB, 256, 0, stream>>>(dst, cnt);

    // ---- layer 1 ----
    zero_f<<<zB, 256, 0, stream>>>(agg, aggN);
    scatter_add<<<sB, 256, 0, stream>>>(x, src, dst, agg);
    sage_gemm_ln<<<gB, 256, 0, stream>>>(x, agg, cnt, wlT1, wrT1,
                                         b1, g1, be1, h1);

    // ---- layer 2 ----
    zero_f<<<zB, 256, 0, stream>>>(agg, aggN);
    scatter_add<<<sB, 256, 0, stream>>>(h1, src, dst, agg);
    sage_gemm_ln<<<gB, 256, 0, stream>>>(h1, agg, cnt, wlT2, wrT2,
                                         b2, g2, be2, out);
}
